// NEATNetwork_7215545057312
// MI455X (gfx1250) — matrix operand order repaired and checked by reading, not compiled, not run on hardware
//
#include <hip/hip_runtime.h>

typedef __attribute__((ext_vector_type(16))) _Float16 v16h;
typedef __attribute__((ext_vector_type(8)))  _Float16 v8h;
typedef __attribute__((ext_vector_type(4)))  _Float16 v4h;
typedef __attribute__((ext_vector_type(8)))  float    v8f;

#define BATCH 2048
#define FAN_IN 32

// ---------------------------------------------------------------------------
// Workspace layout (halfs):
//   actsA : 2048*512            (2 MiB)
//   actsB : 2048*512            (2 MiB)
//   Wslab : dense weights, row-major [N x K] per layer
//     W1 (512x256) @ 0      W2 (512x512) @ 131072   W3 @ 393216
//     W4 (512x512) @ 655360 W5 (64x512)  @ 917504   (total 950272 halfs)
// ---------------------------------------------------------------------------

// Convert fp32 input x [2048 x 256] to f16 activations.
__global__ void k_convert(const float* __restrict__ x, _Float16* __restrict__ a, int n) {
    int i = (blockIdx.x * blockDim.x + threadIdx.x) * 4;
    if (i < n) {
        float4 v = *(const float4*)(x + i);
        v4h h = { (_Float16)v.x, (_Float16)v.y, (_Float16)v.z, (_Float16)v.w };
        *(v4h*)(a + i) = h;
    }
}

// One thread per destination node: zero its dense weight row, then scatter
// its 32 edge weights (duplicate sources accumulate; same thread -> no races).
__global__ void k_densify(const float* __restrict__ w, const int* __restrict__ src,
                          _Float16* __restrict__ wslab) {
    int t = blockIdx.x * blockDim.x + threadIdx.x;
    if (t >= 2112) return;
    int j, Kdim, eBase, srcOff;
    long wOff;
    if (t < 2048) {
        int li = t >> 9;           // 0..3 -> layers 1..4
        j = t & 511;
        Kdim   = (li == 0) ? 256 : 512;
        wOff   = (li == 0) ? 0 : 131072 + (long)(li - 1) * 262144;
        eBase  = li * 16384 + j * 32;
        srcOff = (li == 0) ? 0 : 256 + (li - 1) * 512;  // OFFSETS[li]
    } else {                        // layer 5 (64 nodes)
        j = t - 2048;
        Kdim = 512; wOff = 917504; eBase = 65536 + j * 32; srcOff = 1792;
    }
    _Float16* row = wslab + wOff + (long)j * Kdim;
    uint4 z = make_uint4(0u, 0u, 0u, 0u);
    for (int i = 0; i < Kdim; i += 8) *(uint4*)(row + i) = z;   // zero row (16B stores)
    for (int e = 0; e < FAN_IN; ++e) {
        int s = src[eBase + e] - srcOff;
        row[s] = (_Float16)((float)row[s] + w[eBase + e]);      // sequential RMW
    }
}

// Dense GEMM + sigmoid: out[2048 x N] = sigmoid(acts[2048 x K] * W^T), W is [N x K].
// Block: 256 threads = 8 waves; block tile 128M x 64N; wave tile 16M x 64N.
// B tile is staged memory->LDS with GLOBAL_LOAD_ASYNC_TO_LDS_B128 (ASYNCcnt),
// overlapped with the per-wave A-fragment global loads.
__global__ __launch_bounds__(256)
void k_gemm(const _Float16* __restrict__ acts, const _Float16* __restrict__ W,
            _Float16* __restrict__ outH, float* __restrict__ outF,
            int Kdim, int Ndim, int last) {
    __shared__ alignas(64) _Float16 smem[64 * 32];   // 64(N) x 32(K) weight chunk, 4 KiB

    const int lane  = threadIdx.x & 31;
    const int wave  = threadIdx.x >> 5;
    const int mBase = blockIdx.x * 128 + wave * 16;
    const int nBase = blockIdx.y * 64;
    const int mRow  = mBase + (lane & 15);
    const int sel   = lane >> 4;                     // lane-half selector

    // Per-thread staging slot: 8 halfs (16 B) of the 64x32 tile.
    const int flat = threadIdx.x * 8;
    const int snl  = flat >> 5;                      // local N row   (0..63)
    const int skl  = flat & 31;                      // local K col   (0,8,16,24)
    const unsigned ldsAddr = (unsigned)(unsigned long long)(smem + flat);

    v8f acc[4] = {};                                 // four 16x16 f32 accumulators

    for (int k0 = 0; k0 < Kdim; k0 += 32) {
        // Async copy W[nBase+snl, k0+skl .. +7] (16 B/lane) directly into LDS.
        {
            const _Float16* g = W + (size_t)(nBase + snl) * Kdim + k0 + skl;
            asm volatile("global_load_async_to_lds_b128 %0, %1, off"
                         :: "v"(ldsAddr), "v"((unsigned long long)g)
                         : "memory");
        }
        if (k0 + 32 < Kdim)  // pull next K-chunk of W toward the caches
            __builtin_prefetch(W + (size_t)(nBase + (threadIdx.x >> 2)) * Kdim + k0 + 32, 0, 1);

        // A fragment (16x32 f16), per ISA layout: lane holds M=lane%16,
        // K = 8*sel+{0..7} and 16+8*sel+{0..7} -> two 16B row-major loads.
        // Issued while the async LDS fill is in flight.
        const _Float16* aPtr = acts + (size_t)mRow * Kdim + k0 + sel * 8;
        v8h alo = *(const v8h*)(aPtr);
        v8h ahi = *(const v8h*)(aPtr + 16);
        v16h afrag = __builtin_shufflevector(alo, ahi,
                         0, 1, 2, 3, 4, 5, 6, 7, 8, 9, 10, 11, 12, 13, 14, 15);

        asm volatile("s_wait_asynccnt 0x0" ::: "memory");  // our tile bytes are in LDS
        __syncthreads();                                   // all waves' bytes are in LDS

        // B fragments (32x16 f16): lane holds N=lane%16, K = 8*sel+{0..7} and
        // 16+8*sel+{0..7} -> two 16B slices of the row-major LDS tile.
        #pragma unroll
        for (int t = 0; t < 4; ++t) {
            const _Float16* bPtr = smem + (t * 16 + (lane & 15)) * 32 + sel * 8;
            v8h blo = *(const v8h*)(bPtr);
            v8h bhi = *(const v8h*)(bPtr + 16);
            v16h bfrag = __builtin_shufflevector(blo, bhi,
                         0, 1, 2, 3, 4, 5, 6, 7, 8, 9, 10, 11, 12, 13, 14, 15);
            acc[t] = __builtin_amdgcn_wmma_f32_16x16x32_f16(
                         false, afrag, false, bfrag, (short)0, acc[t], false, false);
        }
        __syncthreads();                                   // tile consumed; safe to refill
    }

    // Epilogue: C/D layout -> row = mBase + r + 8*sel, col = nBase + 16t + lane%16.
    #pragma unroll
    for (int t = 0; t < 4; ++t) {
        int col = nBase + t * 16 + (lane & 15);
        #pragma unroll
        for (int r = 0; r < 8; ++r) {
            int row = mBase + r + sel * 8;
            float s = 1.0f / (1.0f + __expf(-acc[t][r]));
            if (last) outF[(size_t)row * Ndim + col] = s;
            else      outH[(size_t)row * Ndim + col] = (_Float16)s;
        }
    }
}

extern "C" void kernel_launch(void* const* d_in, const int* in_sizes, int n_in,
                              void* d_out, int out_size, void* d_ws, size_t ws_size,
                              hipStream_t stream) {
    const float* x   = (const float*)d_in[0];
    const float* w   = (const float*)d_in[1];
    const int*   src = (const int*)d_in[2];
    // d_in[3] (dst) is implied by the edge ordering; unused.

    _Float16* actsA = (_Float16*)d_ws;
    _Float16* actsB = actsA + 2048 * 512;
    _Float16* wslab = actsB + 2048 * 512;
    float*    out   = (float*)d_out;

    k_convert<<<512, 256, 0, stream>>>(x, actsA, 2048 * 256);
    k_densify<<<(2112 + 255) / 256, 256, 0, stream>>>(w, src, wslab);

    // Layer GEMMs (kernel boundaries provide the inter-layer global sync).
    k_gemm<<<dim3(16, 8), 256, 0, stream>>>(actsA, wslab + 0,      actsB, nullptr, 256, 512, 0);
    k_gemm<<<dim3(16, 8), 256, 0, stream>>>(actsB, wslab + 131072, actsA, nullptr, 512, 512, 0);
    k_gemm<<<dim3(16, 8), 256, 0, stream>>>(actsA, wslab + 393216, actsB, nullptr, 512, 512, 0);
    k_gemm<<<dim3(16, 8), 256, 0, stream>>>(actsB, wslab + 655360, actsA, nullptr, 512, 512, 0);
    k_gemm<<<dim3(16, 1), 256, 0, stream>>>(actsA, wslab + 917504, nullptr, out,   512, 64, 1);
}
